// XdiffTransformer_20744692039772
// MI455X (gfx1250) — matrix operand order repaired and checked by reading, not compiled, not run on hardware
//
#include <hip/hip_runtime.h>
#include <hip/hip_bf16.h>
#include <math.h>

// ---------------------------------------------------------------------------
// Types for CDNA5 WMMA (wave32)
// ---------------------------------------------------------------------------
typedef __attribute__((ext_vector_type(16))) __bf16 v16bf;
typedef __attribute__((ext_vector_type(8)))  __bf16 v8bf;
typedef __attribute__((ext_vector_type(4)))  __bf16 v4bf;
typedef __attribute__((ext_vector_type(8)))  float  v8f;
typedef __attribute__((ext_vector_type(4)))  float  v4f;

#define ACT_NONE     0
#define ACT_GELU     1
#define ACT_SOFTPLUS 2

// Model constants
#define BB    8
#define NN    511
#define SS    512
#define DM    512
#define HH    8
#define DEPTH 64
#define FF    21
#define KK    10
#define DFF   2048

__device__ __forceinline__ float act_apply(float v, int act) {
    if (act == ACT_GELU) {
        return 0.5f * v * (1.0f + erff(v * 0.70710678118654752f));
    } else if (act == ACT_SOFTPLUS) {
        return (v > 20.0f) ? v : log1pf(expf(v));
    }
    return v;
}

__device__ __forceinline__ v16bf join16(v8bf lo, v8bf hi) {
    return __builtin_shufflevector(lo, hi, 0, 1, 2, 3, 4, 5, 6, 7,
                                   8, 9, 10, 11, 12, 13, 14, 15);
}

// ---------------------------------------------------------------------------
// General tiled GEMM: C = act(A[M,K] @ B[K,N] + bias) (+ resid)
// bf16 WMMA 16x16x32, f32 accumulate. Guarded in M, N, K with a vectorized
// fast path for interior tiles. A staged row-major (fragment = two 16B runs),
// B staged TRANSPOSED (fragment = two 16B runs) -> ds_load_b128.
// Next K-step tiles prefetched (global_prefetch_b8) while WMMAs run.
// Block = 256 threads = 8 waves; 64x64 C tile; wave = 16 rows x 32 cols.
// ---------------------------------------------------------------------------
#define BM 64
#define BN 64
#define BK 32
#define ASTR 40   // row stride (bf16): 80 B = multiple of 16, odd bank count

__global__ __launch_bounds__(256)
void k_gemm(const float* __restrict__ A, int lda,
            const float* __restrict__ Bm, int ldb,
            const float* __restrict__ bias,
            const float* __restrict__ resid, int ldr,
            float* __restrict__ C, int ldc,
            int M, int N, int K, int act)
{
    __shared__ __align__(16) __bf16 As[BM][ASTR];   // [m][k]
    __shared__ __align__(16) __bf16 BsT[BN][ASTR];  // [n][k] (transposed)

    const int tid  = threadIdx.x;
    const int wave = tid >> 5;
    const int lane = tid & 31;
    const int half = lane >> 4;
    const int lm   = lane & 15;
    const int wr   = wave >> 1;   // 0..3 : M subtile (16 rows)
    const int wc   = wave & 1;    // 0..1 : N strip (32 cols)
    const int row0 = blockIdx.y * BM;
    const int col0 = blockIdx.x * BN;

    v8f acc0 = {};
    v8f acc1 = {};

    for (int k0 = 0; k0 < K; k0 += BK) {
        const bool fullA = (row0 + BM <= M) && (k0 + BK <= K);
        const bool fullB = (col0 + BN <= N) && (k0 + BK <= K);

        if (fullA) {
            int r = tid >> 2, c8 = (tid & 3) * 8;
            const float* src = A + (size_t)(row0 + r) * lda + (k0 + c8);
            v4f f0 = *(const v4f*)src;
            v4f f1 = *(const v4f*)(src + 4);
            v8bf pk;
            #pragma unroll
            for (int j = 0; j < 4; ++j) {
                pk[j]     = (__bf16)f0[j];
                pk[4 + j] = (__bf16)f1[j];
            }
            *(v8bf*)&As[r][c8] = pk;
        } else {
            #pragma unroll
            for (int i = 0; i < 8; ++i) {
                int li = tid + i * 256;
                int r = li >> 5, c = li & 31;
                int gr = row0 + r, gc = k0 + c;
                float v = (gr < M && gc < K) ? A[(size_t)gr * lda + gc] : 0.0f;
                As[r][c] = (__bf16)v;
            }
        }

        if (fullB) {
            int r = tid >> 3, c8 = (tid & 7) * 8;
            const float* src = Bm + (size_t)(k0 + r) * ldb + (col0 + c8);
            v4f f0 = *(const v4f*)src;
            v4f f1 = *(const v4f*)(src + 4);
            #pragma unroll
            for (int j = 0; j < 4; ++j) {
                BsT[c8 + j][r]     = (__bf16)f0[j];
                BsT[c8 + 4 + j][r] = (__bf16)f1[j];
            }
        } else {
            #pragma unroll
            for (int i = 0; i < 8; ++i) {
                int li = tid + i * 256;
                int r = li >> 6, c = li & 63;
                int gr = k0 + r, gc = col0 + c;
                float v = (gr < K && gc < N) ? Bm[(size_t)gr * ldb + gc] : 0.0f;
                BsT[c][r] = (__bf16)v;
            }
        }

        // prefetch next K-step tiles (global_prefetch_b8, speculative RT)
        int kn = k0 + BK;
        if (kn < K) {
            if (row0 + BM <= M && kn + BK <= K)
                __builtin_prefetch(A + (size_t)(row0 + (tid >> 2)) * lda +
                                   (kn + (tid & 3) * 8), 0, 1);
            if (col0 + BN <= N && kn + BK <= K)
                __builtin_prefetch(Bm + (size_t)(kn + (tid >> 3)) * ldb +
                                   (col0 + (tid & 7) * 8), 0, 1);
        }
        __syncthreads();

        // A fragment: two contiguous 16B runs -> 2x ds_load_b128
        const __bf16* ap = &As[wr * 16 + lm][half * 8];
        v16bf af = join16(*(const v8bf*)ap, *(const v8bf*)(ap + 16));

        // B fragments: two contiguous 16B runs each -> 2x ds_load_b128
        const __bf16* bp0 = &BsT[wc * 32 + lm][half * 8];
        v16bf bf0 = join16(*(const v8bf*)bp0, *(const v8bf*)(bp0 + 16));
        const __bf16* bp1 = &BsT[wc * 32 + 16 + lm][half * 8];
        v16bf bf1 = join16(*(const v8bf*)bp1, *(const v8bf*)(bp1 + 16));

        acc0 = __builtin_amdgcn_wmma_f32_16x16x32_bf16(false, af, false, bf0,
                                                       (short)0, acc0, false, false);
        acc1 = __builtin_amdgcn_wmma_f32_16x16x32_bf16(false, af, false, bf1,
                                                       (short)0, acc1, false, false);
        __syncthreads();
    }

    // epilogue: bias + activation + optional residual
    #pragma unroll
    for (int t = 0; t < 2; ++t) {
        v8f acc = t ? acc1 : acc0;
        int col = col0 + wc * 32 + t * 16 + lm;
        #pragma unroll
        for (int r = 0; r < 8; ++r) {
            int row = row0 + wr * 16 + half * 8 + r;
            if (row < M && col < N) {
                float v = acc[r];
                if (bias) v += bias[col];
                v = act_apply(v, act);
                if (resid) v += resid[(size_t)row * ldr + col];
                C[(size_t)row * ldc + col] = v;
            }
        }
    }
}

// ---------------------------------------------------------------------------
// LayerNorm: one wave per 512-wide row (wave32 shfl reductions)
// ---------------------------------------------------------------------------
__global__ __launch_bounds__(256)
void k_layernorm(const float* __restrict__ in, float* __restrict__ out,
                 const float* __restrict__ g, const float* __restrict__ bt,
                 int rows)
{
    int row  = blockIdx.x * 8 + (threadIdx.x >> 5);
    int lane = threadIdx.x & 31;
    if (row >= rows) return;
    const float* p = in + (size_t)row * DM;
    float s = 0.0f;
    for (int i = lane; i < DM; i += 32) s += p[i];
    #pragma unroll
    for (int off = 16; off > 0; off >>= 1) s += __shfl_xor(s, off, 32);
    float mean = s * (1.0f / DM);
    float v = 0.0f;
    for (int i = lane; i < DM; i += 32) { float d = p[i] - mean; v += d * d; }
    #pragma unroll
    for (int off = 16; off > 0; off >>= 1) v += __shfl_xor(v, off, 32);
    float inv = rsqrtf(v * (1.0f / DM) + 1e-6f);
    float* q = out + (size_t)row * DM;
    for (int i = lane; i < DM; i += 32)
        q[i] = (p[i] - mean) * inv * g[i] + bt[i];
}

// ---------------------------------------------------------------------------
// Fused attention per (b, h, 16-query tile):
//   logits = (Q K^T)/8 + xlog/sqrt(21); softmax; O = P V  (all WMMA bf16)
// K staged row-major (K^T fragment contiguous in depth);
// V staged transposed (P@V fragment contiguous in key) -> ds_load_b128.
// Xk (raw f32 point table) staged via CDNA5 async global->LDS copy.
// LDS total = 65,344 B.
// ---------------------------------------------------------------------------
#define QSTR  72    // Qs / phase-1 KV row stride (bf16): 144 B
#define VSTR  136   // phase-2 V^T row stride (bf16): 272 B
#define LSTR  516   // Lg row stride (f32): 2064 B, lanes hit distinct banks

__global__ __launch_bounds__(256)
void k_attn(const float* __restrict__ qb, const float* __restrict__ kb,
            const float* __restrict__ vb, const float* __restrict__ xlb,
            const float* __restrict__ xall, float* __restrict__ ob)
{
    __shared__ __align__(16) __bf16 Qs[16][QSTR];
    __shared__ __align__(16) float  Xk[SS][3];
    __shared__ float  XLs[16][FF];
    __shared__ __align__(16) __bf16 KVflat[128 * QSTR]; // phase1 [128][72] / phase2 [64][136]
    __shared__ __align__(16) float  Lg[16][LSTR];
    __shared__ float  Opart[4][16][16];                 // also reused as softmax Red[16][16]

    const int bid = blockIdx.x;
    const int qt = bid & 31;
    const int h  = (bid >> 5) & 7;
    const int b  = bid >> 8;
    const int q0 = qt * 16;
    const int tid = threadIdx.x, wave = tid >> 5, lane = tid & 31;
    const int half = lane >> 4, lm = lane & 15;

    // ---- stage Xk: raw f32 copy -> CDNA5 async global->LDS (ASYNCcnt)
    {
        // generic LDS pointer carries the LDS byte offset in its low 32 bits
        unsigned ldsBase = (unsigned)(uintptr_t)&Xk[0][0];
        const float* gsrc = xall + (size_t)b * SS * 3;
        #pragma unroll
        for (int i = 0; i < 6; ++i) {               // 6*256 = 1536 dwords = 512*3
            int idx = tid + i * 256;
            unsigned laddr = ldsBase + (unsigned)idx * 4u;
            unsigned long long gaddr = (unsigned long long)(uintptr_t)(gsrc + idx);
            asm volatile("global_load_async_to_lds_b32 %0, %1, off"
                         :: "v"(laddr), "v"(gaddr) : "memory");
        }
    }

    // ---- stage Q tile (float4 -> v4bf) and xl coefficients
    {
        int r = tid >> 4, c4 = (tid & 15) * 4;
        v4f f = *(const v4f*)(qb + (size_t)(b * SS + q0 + r) * DM + h * DEPTH + c4);
        v4bf pk;
        #pragma unroll
        for (int j = 0; j < 4; ++j) pk[j] = (__bf16)f[j];
        *(v4bf*)&Qs[r][c4] = pk;
    }
    for (int i = tid; i < 16 * FF; i += 256) {
        int r = i / FF, f = i % FF;
        XLs[r][f] = xlb[((size_t)(b * SS + q0 + r)) * (HH * FF) + h * FF + f];
    }
    asm volatile("s_wait_asynccnt 0" ::: "memory");   // own wave's async copies done
    __syncthreads();

    const float inv_dk  = 0.125f;               // 1/sqrt(64)
    const float inv_dkx = 0.21821789023599236f; // 1/sqrt(21)
    const float PI = 3.14159265358979323846f;

    // ---- Phase 1: logits 16 x 512; wave w owns key-cols [kt+w*16, +16)
    for (int kt = 0; kt < SS; kt += 128) {
        #pragma unroll
        for (int i = 0; i < 8; ++i) {
            int li = tid + i * 256;
            int r = li >> 4, c4 = (li & 15) * 4;
            v4f f = *(const v4f*)(kb + (size_t)(b * SS + kt + r) * DM + h * DEPTH + c4);
            v4bf pk;
            #pragma unroll
            for (int j = 0; j < 4; ++j) pk[j] = (__bf16)f[j];
            *(v4bf*)&KVflat[r * QSTR + c4] = pk;
        }
        __syncthreads();

        v8f acc = {};
        #pragma unroll
        for (int d0 = 0; d0 < DEPTH; d0 += 32) {
            const __bf16* ap = &Qs[lm][d0 + half * 8];
            v16bf af = join16(*(const v8bf*)ap, *(const v8bf*)(ap + 16));
            const __bf16* bp = &KVflat[(wave * 16 + lm) * QSTR + d0 + half * 8];
            v16bf bfv = join16(*(const v8bf*)bp, *(const v8bf*)(bp + 16));
            acc = __builtin_amdgcn_wmma_f32_16x16x32_bf16(false, af, false, bfv,
                                                          (short)0, acc, false, false);
        }
        // fuse xlog (distance + sin/cos series) into the accumulator
        int kcol = kt + wave * 16 + lm;
        float xk0 = Xk[kcol][0], xk1 = Xk[kcol][1], xk2 = Xk[kcol][2];
        #pragma unroll
        for (int r = 0; r < 8; ++r) {
            int m = half * 8 + r;
            float dx = Xk[q0 + m][0] - xk0;
            float dy = Xk[q0 + m][1] - xk1;
            float dz = Xk[q0 + m][2] - xk2;
            float rr = sqrtf(dx * dx + dy * dy + dz * dz);
            float xlog = rr * XLs[m][0];
            #pragma unroll
            for (int i = 1; i <= KK; ++i) {
                float ang = rr * (PI * (float)i);
                xlog += __sinf(ang) * XLs[m][2 * i - 1] + __cosf(ang) * XLs[m][2 * i];
            }
            Lg[m][kcol] = acc[r] * inv_dk + xlog * inv_dkx;
        }
        __syncthreads();
    }

    // ---- Softmax over 512 keys (16 threads per row); Red aliases Opart
    {
        float (*Red)[16] = (float (*)[16]) & Opart[0][0][0];
        int row = tid >> 4;
        int sub = tid & 15;
        float mx = -1e30f;
        for (int c = sub; c < SS; c += 16) mx = fmaxf(mx, Lg[row][c]);
        Red[row][sub] = mx;
        __syncthreads();
        if (sub == 0) {
            float m2 = Red[row][0];
            #pragma unroll
            for (int i = 1; i < 16; ++i) m2 = fmaxf(m2, Red[row][i]);
            Red[row][0] = m2;
        }
        __syncthreads();
        mx = Red[row][0];
        __syncthreads();
        float sum = 0.0f;
        for (int c = sub; c < SS; c += 16) {
            float e = __expf(Lg[row][c] - mx);
            Lg[row][c] = e;
            sum += e;
        }
        Red[row][sub] = sum;
        __syncthreads();
        if (sub == 0) {
            float s2 = 0.0f;
            #pragma unroll
            for (int i = 0; i < 16; ++i) s2 += Red[row][i];
            Red[row][0] = s2;
        }
        __syncthreads();
        float inv = 1.0f / Red[row][0];
        for (int c = sub; c < SS; c += 16) Lg[row][c] *= inv;
        __syncthreads();
    }

    // ---- Phase 2: O = P @ V, split-K across wave pairs, LDS reduce
    const int tile  = wave & 3;   // depth 16-wide output tile
    const int khalf = wave >> 2;  // which half of the 512 keys
    v8f oacc = {};
    for (int kt = 0; kt < SS; kt += 128) {
        // stage V transposed: KVflat[d * VSTR + key]
        #pragma unroll
        for (int i = 0; i < 8; ++i) {
            int li = tid + i * 256;
            int r = li >> 4, c4 = (li & 15) * 4;   // r = key, c4 = depth
            v4f f = *(const v4f*)(vb + (size_t)(b * SS + kt + r) * DM + h * DEPTH + c4);
            #pragma unroll
            for (int j = 0; j < 4; ++j)
                KVflat[(c4 + j) * VSTR + r] = (__bf16)f[j];
        }
        __syncthreads();
        if ((kt >> 8) == khalf) {
            #pragma unroll
            for (int kk0 = 0; kk0 < 128; kk0 += 32) {
                // A fragment from probabilities: two aligned 32B f32 runs
                const float* lp = &Lg[lm][kt + kk0 + half * 8];
                v4f a0 = *(const v4f*)lp;
                v4f a1 = *(const v4f*)(lp + 4);
                v4f a2 = *(const v4f*)(lp + 16);
                v4f a3 = *(const v4f*)(lp + 20);
                v16bf af;
                #pragma unroll
                for (int j = 0; j < 4; ++j) {
                    af[j]      = (__bf16)a0[j];
                    af[4 + j]  = (__bf16)a1[j];
                    af[8 + j]  = (__bf16)a2[j];
                    af[12 + j] = (__bf16)a3[j];
                }
                const __bf16* bp = &KVflat[(tile * 16 + lm) * VSTR + kk0 + half * 8];
                v16bf bfv = join16(*(const v8bf*)bp, *(const v8bf*)(bp + 16));
                oacc = __builtin_amdgcn_wmma_f32_16x16x32_bf16(false, af, false, bfv,
                                                               (short)0, oacc, false, false);
            }
        }
        __syncthreads();
    }
    if (khalf == 1) {
        #pragma unroll
        for (int r = 0; r < 8; ++r)
            Opart[tile][half * 8 + r][lm] = oacc[r];
    }
    __syncthreads();
    if (khalf == 0) {
        #pragma unroll
        for (int r = 0; r < 8; ++r) {
            int m = half * 8 + r;
            ob[((size_t)(b * SS + q0 + m)) * DM + h * DEPTH + tile * 16 + lm] =
                oacc[r] + Opart[tile][m][lm];
        }
    }
}

// ---------------------------------------------------------------------------
// Small elementwise helpers
// ---------------------------------------------------------------------------
__global__ void k_build_xall(const float* __restrict__ x,
                             const float* __restrict__ xin,
                             float* __restrict__ xall)
{
    int i = blockIdx.x * 256 + threadIdx.x;   // BB*SS*3
    if (i >= BB * SS * 3) return;
    int d = i % 3;
    int s = (i / 3) % SS;
    int b = i / (3 * SS);
    xall[i] = (s == 0) ? x[b * 3 + d]
                       : xin[((size_t)b * NN + (s - 1)) * 3 + d];
}

__global__ void k_build_latents(const float* __restrict__ tok,
                                const float* __restrict__ lat,
                                float* __restrict__ latents)
{
    int i = blockIdx.x * 256 + threadIdx.x;   // BB*SS*DM
    if (i >= BB * SS * DM) return;
    int d = i & (DM - 1);
    int s = (i >> 9) & (SS - 1);
    int b = i >> 18;
    latents[i] = (s == 0) ? tok[d]
                          : lat[((size_t)(b * NN + s - 1)) * DM + d];
}

// ---------------------------------------------------------------------------
// Host launcher
// ---------------------------------------------------------------------------
static inline dim3 gemm_grid(int M, int N) {
    return dim3((N + BN - 1) / BN, (M + BM - 1) / BM);
}

extern "C" void kernel_launch(void* const* d_in, const int* in_sizes, int n_in,
                              void* d_out, int out_size, void* d_ws, size_t ws_size,
                              hipStream_t stream)
{
    (void)in_sizes; (void)n_in; (void)out_size; (void)ws_size;
    auto P = [&](int i) { return (const float*)d_in[i]; };

    // Flattened parameter indices (reference dict insertion order):
    // 0:x 1:x_inputs 2:inputs 3:Win0 4:bin0 5:Win1 6:bin1 7:x_token
    // per layer l (base = 8 + 18*l):
    //   +0 ln1.g +1 ln1.b +2 ln2.g +3 ln2.b +4 wq.W +5 wq.b +6 wk.W +7 wk.b
    //   +8 wv.W +9 wv.b +10 wx.W +11 wx.b +12 wo.W +13 wo.b
    //   +14 ff1.W +15 ff1.b +16 ff2.W +17 ff2.b
    // 80 ln_f.g 81 ln_f.b 82 pre.W 83 pre.b 84 fin.W 85 fin.b

    const int M  = BB * SS;      // 4096
    const int Mn = BB * NN;      // 4088

    // Workspace layout (floats)
    float* ws = (float*)d_ws;
    size_t o = 0;
    float* xall    = ws + o; o += (size_t)BB * SS * 3;
    o = (o + 255) & ~(size_t)255;
    float* latents = ws + o; o += (size_t)M * DM;
    float* lnb     = ws + o; o += (size_t)M * DM;
    float* qbuf    = ws + o; o += (size_t)M * DM;
    float* kbuf    = ws + o; o += (size_t)M * DM;
    float* vbuf    = ws + o; o += (size_t)M * DM;
    float* obuf    = ws + o; o += (size_t)M * DM;
    float* xlbuf   = ws + o; o += (size_t)M * (HH * FF);
    o = (o + 255) & ~(size_t)255;
    float* ffm     = ws + o; o += (size_t)M * DFF;      // also aliases tmp1/tmp2
    float* ttok    = ws + o; o += (size_t)BB * DM;
    float* tmp1 = ffm;                       // Mn x 512
    float* tmp2 = ffm + (size_t)Mn * DM;     // Mn x 512 (fits: 2*Mn*512 < M*DFF)

    // x_all
    k_build_xall<<<(BB * SS * 3 + 255) / 256, 256, 0, stream>>>(P(0), P(1), xall);

    // input MLP: softplus(inputs @ Win0 + b) -> gelu(. @ Win1 + b)
    k_gemm<<<gemm_grid(Mn, DM), 256, 0, stream>>>(P(2), 8,  P(3), DM, P(4),
                                                  nullptr, 0, tmp1, DM,
                                                  Mn, DM, 8, ACT_SOFTPLUS);
    k_gemm<<<gemm_grid(Mn, DM), 256, 0, stream>>>(tmp1, DM, P(5), DM, P(6),
                                                  nullptr, 0, tmp2, DM,
                                                  Mn, DM, DM, ACT_GELU);
    k_build_latents<<<(M * DM + 255) / 256, 256, 0, stream>>>(P(7), tmp2, latents);

    for (int l = 0; l < 4; ++l) {
        int base = 8 + 18 * l;
        // ln1 -> q,k,v,xl
        k_layernorm<<<M / 8, 256, 0, stream>>>(latents, lnb, P(base + 0), P(base + 1), M);
        k_gemm<<<gemm_grid(M, DM), 256, 0, stream>>>(lnb, DM, P(base + 4), DM, P(base + 5),
                                                     nullptr, 0, qbuf, DM, M, DM, DM, ACT_NONE);
        k_gemm<<<gemm_grid(M, DM), 256, 0, stream>>>(lnb, DM, P(base + 6), DM, P(base + 7),
                                                     nullptr, 0, kbuf, DM, M, DM, DM, ACT_NONE);
        k_gemm<<<gemm_grid(M, DM), 256, 0, stream>>>(lnb, DM, P(base + 8), DM, P(base + 9),
                                                     nullptr, 0, vbuf, DM, M, DM, DM, ACT_NONE);
        k_gemm<<<gemm_grid(M, HH * FF), 256, 0, stream>>>(qbuf, DM, P(base + 10), HH * FF,
                                                          P(base + 11), nullptr, 0,
                                                          xlbuf, HH * FF, M, HH * FF, DM, ACT_NONE);
        // fused attention (b,h,q-tile): 8*8*32 workgroups
        k_attn<<<BB * HH * (SS / 16), 256, 0, stream>>>(qbuf, kbuf, vbuf, xlbuf, xall, obuf);
        // wo projection + residual (in place on latents)
        k_gemm<<<gemm_grid(M, DM), 256, 0, stream>>>(obuf, DM, P(base + 12), DM, P(base + 13),
                                                     latents, DM, latents, DM, M, DM, DM, ACT_NONE);
        // FFN
        k_layernorm<<<M / 8, 256, 0, stream>>>(latents, lnb, P(base + 2), P(base + 3), M);
        k_gemm<<<gemm_grid(M, DFF), 256, 0, stream>>>(lnb, DM, P(base + 14), DFF, P(base + 15),
                                                      nullptr, 0, ffm, DFF, M, DFF, DM, ACT_GELU);
        k_gemm<<<gemm_grid(M, DM), 256, 0, stream>>>(ffm, DFF, P(base + 16), DM, P(base + 17),
                                                     latents, DM, latents, DM, M, DM, DFF, ACT_NONE);
    }

    // final LN, token-0 extraction (strided lda), head MLP
    k_layernorm<<<M / 8, 256, 0, stream>>>(latents, lnb, P(80), P(81), M);
    k_gemm<<<gemm_grid(BB, DM), 256, 0, stream>>>(lnb, SS * DM, P(82), DM, P(83),
                                                  nullptr, 0, ttok, DM, BB, DM, DM, ACT_GELU);
    k_gemm<<<gemm_grid(BB, 1), 256, 0, stream>>>(ttok, DM, P(84), 1, P(85),
                                                 nullptr, 0, (float*)d_out, 1,
                                                 BB, 1, DM, ACT_NONE);
}
